// ContrastiveLoss_15753940041890
// MI455X (gfx1250) — compile-verified, matrix operand order checked by reading, not run on hardware
//
#include <hip/hip_runtime.h>

// ---------------------------------------------------------------------------
// MI455X / gfx1250 contrastive-loss kernel.
// Compute-bound: ~52 GFLOP of GEMM vs ~24MB of data (L2-resident).
// Uses split-f16 (hi/lo) WMMA: A.B ~= Ah.Bh + Ah.Bl + Al.Bh, f32 accumulate
// via v_wmma_f32_16x16x32_f16 -> ~22-bit effective mantissa.
// ---------------------------------------------------------------------------

typedef __attribute__((ext_vector_type(16))) _Float16 v16h;
typedef __attribute__((ext_vector_type(8)))  _Float16 v8h;
typedef __attribute__((ext_vector_type(8)))  float    v8f;

#define INV_T (1.0f / 0.07f)
#define DIM   256

// Load a 16-half A/B fragment for v_wmma_f32_16x16x32_f16 from a row-major
// [row][256] f16 matrix. Per ISA 16-bit A layout (16x32, MxK):
//   lane L: m = L&15, koff = (L>>4)*8
//   halves 0..7  = K = ks*32 + koff + 0..7
//   halves 8..15 = K = ks*32 + 16 + koff + 0..7
// B (32x16, KxN) for sim = X @ X^T has B[k][n] = X[colbase+n][k] -> identical
// per-lane addressing with row = colbase + (L&15).
__device__ __forceinline__ v16h load_frag16(const _Float16* __restrict__ row,
                                            int ks, int koff) {
  const v8h a0 = *reinterpret_cast<const v8h*>(row + ks * 32 + koff);
  const v8h a1 = *reinterpret_cast<const v8h*>(row + ks * 32 + 16 + koff);
  return __builtin_shufflevector(a0, a1, 0, 1, 2, 3, 4, 5, 6, 7,
                                 8, 9, 10, 11, 12, 13, 14, 15);
}

// ---------------------------------------------------------------------------
// Row L2-normalize + split into f16 hi/lo. One wave per 256-elem row.
// ---------------------------------------------------------------------------
__global__ __launch_bounds__(256) void norm_split(const float* __restrict__ src,
                                                  _Float16* __restrict__ hi,
                                                  _Float16* __restrict__ lo,
                                                  int rows) {
  int gid  = blockIdx.x * blockDim.x + threadIdx.x;
  int wave = gid >> 5;
  int lane = gid & 31;
  if (wave >= rows) return;

  const float* r = src + (size_t)wave * DIM + lane * 8;
  float x[8];
  float ss = 0.f;
#pragma unroll
  for (int i = 0; i < 8; ++i) { x[i] = r[i]; ss += x[i] * x[i]; }
#pragma unroll
  for (int m = 16; m >= 1; m >>= 1) ss += __shfl_xor(ss, m, 32);

  float inv = 1.0f / fmaxf(sqrtf(ss), 1e-12f);
  _Float16* ph = hi + (size_t)wave * DIM + lane * 8;
  _Float16* pl = lo + (size_t)wave * DIM + lane * 8;
#pragma unroll
  for (int i = 0; i < 8; ++i) {
    float v     = x[i] * inv;
    _Float16 h  = (_Float16)v;
    ph[i] = h;
    pl[i] = (_Float16)(v - (float)h);
  }
}

// ---------------------------------------------------------------------------
// WMMA similarity sweep. One 32-thread wave per block.
// Block (bx, by): rows [bx*16, bx*16+16), column tiles [by*TPC, (by+1)*TPC).
// MODE 0 (SimCLR): A rows direct; record diag (slotQ) and pair (slotP) sims.
// MODE 1 (spatial): A rows gathered via aidx; record anchor (slotP) and
//                   neighbor (slotQ) column sims.
// Per-row sum of exp(sim/T) accumulated via shfl butterfly + atomicAdd.
// ---------------------------------------------------------------------------
template <int MODE>
__global__ __launch_bounds__(32) void contrast_gemm(
    const _Float16* __restrict__ Ahi_m, const _Float16* __restrict__ Alo_m,
    const _Float16* __restrict__ Bhi_m, const _Float16* __restrict__ Blo_m,
    const int* __restrict__ aidx, const int* __restrict__ nidx,
    float* __restrict__ rowsum, float* __restrict__ slotP,
    float* __restrict__ slotQ, int tilesPerChunk, int nB) {
  const int lane    = threadIdx.x & 31;
  const int rowbase = blockIdx.x * 16;
  const int t0      = blockIdx.y * tilesPerChunk;
  const int koff    = (lane >> 4) * 8;
  const int mA      = lane & 15;  // A row / B column within tile

  // --- preload A fragments for full K=256 (8 k-steps, hi+lo) ---
  int arow = rowbase + mA;
  size_t asrc = (MODE == 1) ? (size_t)aidx[arow] : (size_t)arow;
  const _Float16* ahp = Ahi_m + asrc * DIM;
  const _Float16* alp = Alo_m + asrc * DIM;

  v16h Ah[8], Al[8];
#pragma unroll
  for (int ks = 0; ks < 8; ++ks) {
    Ah[ks] = load_frag16(ahp, ks, koff);
    Al[ks] = load_frag16(alp, ks, koff);
  }

  // C-layout rows owned by this lane: rbase2 + v, v = 0..7
  const int rbase2 = rowbase + 8 * (lane >> 4);
  int pcol[8], qcol[8];
#pragma unroll
  for (int v = 0; v < 8; ++v) {
    int i = rbase2 + v;
    if (MODE == 0) {
      qcol[v] = i;                               // diagonal column
      pcol[v] = (i < nB) ? i + nB : i - nB;      // positive-pair column
    } else {
      pcol[v] = aidx[i];                         // masked anchor column
      qcol[v] = nidx[i];                         // masked neighbor column
    }
  }

  v8f rs = {0.f, 0.f, 0.f, 0.f, 0.f, 0.f, 0.f, 0.f};

  for (int t = 0; t < tilesPerChunk; ++t) {
    const int colbase = (t0 + t) * 16;
    const _Float16* bhp = Bhi_m + (size_t)(colbase + mA) * DIM;
    const _Float16* blp = Blo_m + (size_t)(colbase + mA) * DIM;

    v8f acc = {0.f, 0.f, 0.f, 0.f, 0.f, 0.f, 0.f, 0.f};
#pragma unroll
    for (int ks = 0; ks < 8; ++ks) {
      v16h Bh = load_frag16(bhp, ks, koff);
      v16h Bl = load_frag16(blp, ks, koff);
      acc = __builtin_amdgcn_wmma_f32_16x16x32_f16(false, Ah[ks], false, Bh,
                                                   (short)0, acc, false, false);
      acc = __builtin_amdgcn_wmma_f32_16x16x32_f16(false, Ah[ks], false, Bl,
                                                   (short)0, acc, false, false);
      acc = __builtin_amdgcn_wmma_f32_16x16x32_f16(false, Al[ks], false, Bh,
                                                   (short)0, acc, false, false);
    }

    const int j = colbase + mA;  // this lane's column
#pragma unroll
    for (int v = 0; v < 8; ++v) {
      float s = acc[v] * INV_T;
      rs[v] += __expf(s);        // v_exp_f32 (TRANS) co-executes with WMMA
      if (j == qcol[v]) slotQ[rbase2 + v] = s;
      if (j == pcol[v]) slotP[rbase2 + v] = s;
    }
  }

  // reduce across the 16 lanes sharing each row set (masks 8..1 stay in-half)
#pragma unroll
  for (int m = 8; m >= 1; m >>= 1) {
#pragma unroll
    for (int v = 0; v < 8; ++v) rs[v] += __shfl_xor(rs[v], m, 32);
  }
  if ((lane & 15) == 0) {
#pragma unroll
    for (int v = 0; v < 8; ++v) atomicAdd(&rowsum[rbase2 + v], rs[v]);
  }
}

// ---------------------------------------------------------------------------
// pos[p] = <e_norm[a_p], e_norm[n_p]> / T, computed in f32 from raw embeddings.
// One wave per pair.
// ---------------------------------------------------------------------------
__global__ __launch_bounds__(256) void pos_kernel(const float* __restrict__ emb,
                                                  const int* __restrict__ aidx,
                                                  const int* __restrict__ nidx,
                                                  float* __restrict__ pos,
                                                  int P) {
  int gid  = blockIdx.x * blockDim.x + threadIdx.x;
  int wave = gid >> 5;
  int lane = gid & 31;
  if (wave >= P) return;

  const float* ea = emb + (size_t)aidx[wave] * DIM + lane * 8;
  const float* en = emb + (size_t)nidx[wave] * DIM + lane * 8;
  float d = 0.f, sa = 0.f, sn = 0.f;
#pragma unroll
  for (int i = 0; i < 8; ++i) {
    float xa = ea[i], xn = en[i];
    d += xa * xn; sa += xa * xa; sn += xn * xn;
  }
#pragma unroll
  for (int m = 16; m >= 1; m >>= 1) {
    d  += __shfl_xor(d, m, 32);
    sa += __shfl_xor(sa, m, 32);
    sn += __shfl_xor(sn, m, 32);
  }
  if (lane == 0)
    pos[wave] = d / (fmaxf(sqrtf(sa), 1e-12f) * fmaxf(sqrtf(sn), 1e-12f)) * INV_T;
}

// ---------------------------------------------------------------------------
// Finalize reductions (single block each, deterministic shared-mem tree).
// ---------------------------------------------------------------------------
__global__ __launch_bounds__(256) void finalize_simclr(
    const float* __restrict__ rowsum, const float* __restrict__ diag,
    const float* __restrict__ pair, float* __restrict__ out, int n) {
  __shared__ float sh[256];
  float s = 0.f;
  for (int i = threadIdx.x; i < n; i += 256) {
    float sum_exp = rowsum[i] - __expf(diag[i]);  // exact cancel of diag term
    s += __logf(sum_exp) - pair[i];
  }
  sh[threadIdx.x] = s;
  __syncthreads();
  for (int st = 128; st > 0; st >>= 1) {
    if (threadIdx.x < st) sh[threadIdx.x] += sh[threadIdx.x + st];
    __syncthreads();
  }
  if (threadIdx.x == 0) out[0] = sh[0] / (float)n;
}

__global__ __launch_bounds__(256) void finalize_spatial(
    const float* __restrict__ rowsum, const float* __restrict__ sa,
    const float* __restrict__ sn, const float* __restrict__ pos,
    const int* __restrict__ aidx, const int* __restrict__ nidx,
    float* __restrict__ out, int P) {
  __shared__ float sh[256];
  float s = 0.f;
  for (int p = threadIdx.x; p < P; p += 256) {
    float S = __expf(pos[p]) + rowsum[p] - __expf(sa[p]);
    if (aidx[p] != nidx[p]) S -= __expf(sn[p]);
    s += __logf(S) - pos[p];
  }
  sh[threadIdx.x] = s;
  __syncthreads();
  for (int st = 128; st > 0; st >>= 1) {
    if (threadIdx.x < st) sh[threadIdx.x] += sh[threadIdx.x + st];
    __syncthreads();
  }
  if (threadIdx.x == 0) out[1] = sh[0] / (float)P;
}

// ---------------------------------------------------------------------------
extern "C" void kernel_launch(void* const* d_in, const int* in_sizes, int n_in,
                              void* d_out, int out_size, void* d_ws,
                              size_t ws_size, hipStream_t stream) {
  const float* z1  = (const float*)d_in[0];  // [4096, 256]
  const float* z2  = (const float*)d_in[1];  // [4096, 256]
  const float* emb = (const float*)d_in[2];  // [8192, 256]
  const int* aidx  = (const int*)d_in[3];    // [4096]
  const int* nidx  = (const int*)d_in[4];    // [4096]
  float* out       = (float*)d_out;          // [2]

  const int Bn = 4096, Nn = 8192, Pp = 4096;
  const int N2 = 2 * Bn;  // 8192 SimCLR rows

  // ---- workspace layout ----
  _Float16* z_hi = (_Float16*)d_ws;                 // [8192*256] f16
  _Float16* z_lo = z_hi + (size_t)N2 * DIM;
  _Float16* e_hi = z_lo + (size_t)N2 * DIM;
  _Float16* e_lo = e_hi + (size_t)Nn * DIM;
  float* rowsum1 = (float*)(e_lo + (size_t)Nn * DIM);  // [8192]
  float* diag1   = rowsum1 + N2;                        // [8192]
  float* pair1   = diag1 + N2;                          // [8192]
  float* rowsum2 = pair1 + N2;                          // [4096]
  float* sa      = rowsum2 + Pp;                        // [4096]
  float* sn      = sa + Pp;                             // [4096]
  float* pos     = sn + Pp;                             // [4096]

  // zero accumulators (capture-safe memset nodes)
  hipMemsetAsync(rowsum1, 0, N2 * sizeof(float), stream);
  hipMemsetAsync(rowsum2, 0, Pp * sizeof(float), stream);

  // ---- normalize + hi/lo split (8 rows per 256-thread block) ----
  norm_split<<<Bn / 8, 256, 0, stream>>>(z1, z_hi, z_lo, Bn);
  norm_split<<<Bn / 8, 256, 0, stream>>>(z2, z_hi + (size_t)Bn * DIM,
                                         z_lo + (size_t)Bn * DIM, Bn);
  norm_split<<<Nn / 8, 256, 0, stream>>>(emb, e_hi, e_lo, Nn);

  // ---- WMMA sweeps ----
  // SimCLR: 512 row tiles x 16 chunks of 32 column tiles (512 tiles = 8192 cols)
  contrast_gemm<0><<<dim3(N2 / 16, 16), 32, 0, stream>>>(
      z_hi, z_lo, z_hi, z_lo, nullptr, nullptr, rowsum1, pair1, diag1, 32, Bn);
  // Spatial: 256 row tiles (gathered anchors) x 16 chunks of 32 column tiles
  contrast_gemm<1><<<dim3(Pp / 16, 16), 32, 0, stream>>>(
      e_hi, e_lo, e_hi, e_lo, aidx, nidx, rowsum2, sa, sn, 32, 0);

  // ---- positives + finalize ----
  pos_kernel<<<(Pp * 32) / 256, 256, 0, stream>>>(emb, aidx, nidx, pos, Pp);
  finalize_simclr<<<1, 256, 0, stream>>>(rowsum1, diag1, pair1, out, N2);
  finalize_spatial<<<1, 256, 0, stream>>>(rowsum2, sa, sn, pos, aidx, nidx, out,
                                          Pp);
}
